// MOELayer_13743895347201
// MI455X (gfx1250) — matrix operand order repaired and checked by reading, not compiled, not run on hardware
//
#include <hip/hip_runtime.h>
#include <math.h>

typedef __attribute__((ext_vector_type(16))) _Float16 v16h;
typedef __attribute__((ext_vector_type(8)))  _Float16 v8h;
typedef __attribute__((ext_vector_type(4)))  _Float16 v4h;
typedef __attribute__((ext_vector_type(8)))  float    v8f;
typedef __attribute__((ext_vector_type(4)))  int      v4i;

#ifndef __has_builtin
#define __has_builtin(x) 0
#endif
#if __has_builtin(__builtin_amdgcn_global_load_async_to_lds_b128) && \
    __has_builtin(__builtin_amdgcn_s_wait_asynccnt)
#define HAVE_ASYNC 1
#else
#define HAVE_ASYNC 0
#endif

#define GAS __attribute__((address_space(1)))
#define LAS __attribute__((address_space(3)))

constexpr int Bq = 2, Sq = 4096, Mq = 1024, Eq = 16, Kq = 2, Hq = 2048;
constexpr int Nq = Bq * Sq;     // 8192 tokens
constexpr int Cq = 1024;        // capacity per expert (K*N/E)

// ---------------------------------------------------------------- gating ----
__global__ __launch_bounds__(128) void gate_kernel(
    const float* __restrict__ x, const float* __restrict__ wg,
    int* __restrict__ top_i, float* __restrict__ gates) {
  const int n = blockIdx.x;
  const int t = threadIdx.x;
  __shared__ float red[128][16];

  float acc[16];
#pragma unroll
  for (int e = 0; e < 16; ++e) acc[e] = 0.f;

  const float* xr = x + (size_t)n * Mq;
  for (int m = t; m < Mq; m += 128) {
    const float xv = xr[m];
    const float* wr = wg + m * Eq;   // contiguous 16 floats
#pragma unroll
    for (int e = 0; e < 16; ++e) acc[e] += xv * wr[e];
  }
#pragma unroll
  for (int e = 0; e < 16; ++e) red[t][e] = acc[e];
  __syncthreads();
  for (int s = 64; s > 0; s >>= 1) {
    if (t < s) {
#pragma unroll
      for (int e = 0; e < 16; ++e) red[t][e] += red[t + s][e];
    }
    __syncthreads();
  }
  if (t == 0) {
    float l[16];
#pragma unroll
    for (int e = 0; e < 16; ++e) l[e] = red[0][e];
    float mx = l[0];
#pragma unroll
    for (int e = 1; e < 16; ++e) mx = fmaxf(mx, l[e]);
    float p[16];
#pragma unroll
    for (int e = 0; e < 16; ++e) p[e] = __expf(l[e] - mx);
    // top-1 (first occurrence on ties, matches lax.top_k)
    int i0 = 0; float b0 = p[0];
#pragma unroll
    for (int e = 1; e < 16; ++e) { if (p[e] > b0) { b0 = p[e]; i0 = e; } }
    // top-2
    int i1 = -1; float b1v = -1.f;
#pragma unroll
    for (int e = 0; e < 16; ++e) {
      if (e != i0 && p[e] > b1v) { b1v = p[e]; i1 = e; }
    }
    const float inv = 1.f / (b0 + b1v);
    top_i[n * 2 + 0] = i0;
    top_i[n * 2 + 1] = i1;
    gates[n * 2 + 0] = b0 * inv;
    gates[n * 2 + 1] = b1v * inv;
  }
}

// ------------------------------------------- deterministic capacity scan ----
// GShard order: flat index j = k*N + n over K*N = 16384 entries.
// One block, 1024 threads x 16 entries each; Hillis-Steele scan of 16-vectors.
__global__ __launch_bounds__(1024) void scan_kernel(
    const int* __restrict__ top_i, int* __restrict__ slots) {
  __shared__ int lds[1024 * 16];   // 64 KB
  const int t = threadIdx.x;
  const int j0 = t * 16;

  int ex[16];
#pragma unroll
  for (int i = 0; i < 16; ++i) {
    const int j = j0 + i;
    const int k = j >> 13;         // j / N
    const int n = j & (Nq - 1);
    ex[i] = top_i[n * 2 + k];
  }
  int cnt[16];
#pragma unroll
  for (int e = 0; e < 16; ++e) {
    int c = 0;
#pragma unroll
    for (int i = 0; i < 16; ++i) c += (ex[i] == e) ? 1 : 0;
    cnt[e] = c;
    lds[t * 16 + e] = c;
  }
  __syncthreads();
  for (int s = 1; s < 1024; s <<= 1) {
    int tmp[16];
#pragma unroll
    for (int e = 0; e < 16; ++e) tmp[e] = 0;
    if (t >= s) {
#pragma unroll
      for (int e = 0; e < 16; ++e) tmp[e] = lds[(t - s) * 16 + e];
    }
    __syncthreads();
#pragma unroll
    for (int e = 0; e < 16; ++e) lds[t * 16 + e] += tmp[e];
    __syncthreads();
  }
  // make my slot exclusive (inclusive - own)
#pragma unroll
  for (int e = 0; e < 16; ++e) lds[t * 16 + e] -= cnt[e];

#pragma unroll
  for (int i = 0; i < 16; ++i) {
    const int j = j0 + i;
    const int k = j >> 13;
    const int n = j & (Nq - 1);
    const int e = ex[i];
    int prior = 0;
#pragma unroll
    for (int i2 = 0; i2 < 16; ++i2)
      if (i2 < i) prior += (ex[i2] == e) ? 1 : 0;
    const int pos = lds[t * 16 + e] + prior;   // dynamic LDS index: fine
    slots[n * 2 + k] = (pos < Cq) ? (e * Cq + pos) : -1;
  }
}

// --------------------------------------------------------------- dispatch ---
__global__ __launch_bounds__(256) void dispatch_kernel(
    const float* __restrict__ x, const int* __restrict__ slots,
    _Float16* __restrict__ disp) {
  const int j = blockIdx.x;            // (n,k)
  const int n = j >> 1, k = j & 1;
  const int slot = slots[n * 2 + k];
  if (slot < 0) return;
  const float4* src = (const float4*)(x + (size_t)n * Mq);
  v4h* dst = (v4h*)(disp + (size_t)slot * Mq);
  for (int i = threadIdx.x; i < Mq / 4; i += 256) {
    const float4 v = src[i];
    v4h h;
    h[0] = (_Float16)v.x; h[1] = (_Float16)v.y;
    h[2] = (_Float16)v.z; h[3] = (_Float16)v.w;
    dst[i] = h;
  }
}

// ------------------------------------------------------------ WMMA GEMM -----
// Per expert:  Out[C x Nd] = act( A[C x Kd] (f16) * B[Kd x Nd] (f32) + bias )
// Block tile 128x128, BK=32, 8 waves (4x2), wave tile 32x64 = 2x4 WMMA tiles.
// Double-buffered LDS pipeline; A operand streamed with async global->LDS
// (ASYNCcnt) when the toolchain exposes the gfx1250 builtins.
constexpr int BM = 128, BN = 128, BK = 32;
constexpr int LS = 40;   // LDS row stride in halves (bank-spread padding)

__device__ __forceinline__ void loadB_regs(const float* __restrict__ Be,
                                           int Nd, int k0, int tid,
                                           float4 (&bv)[4]) {
#pragma unroll
  for (int i = 0; i < 4; ++i) {
    const int idx = tid + i * 256;
    const int kr = idx >> 5, n4 = (idx & 31) * 4;
    bv[i] = *(const float4*)(Be + (size_t)(k0 + kr) * Nd + n4);
  }
}

__device__ __forceinline__ void storeB_lds(_Float16* __restrict__ Bs, int tid,
                                           const float4 (&bv)[4]) {
#pragma unroll
  for (int i = 0; i < 4; ++i) {
    const int idx = tid + i * 256;
    const int kr = idx >> 5, n4 = (idx & 31) * 4;
    Bs[(n4 + 0) * LS + kr] = (_Float16)bv[i].x;
    Bs[(n4 + 1) * LS + kr] = (_Float16)bv[i].y;
    Bs[(n4 + 2) * LS + kr] = (_Float16)bv[i].z;
    Bs[(n4 + 3) * LS + kr] = (_Float16)bv[i].w;
  }
}

#if HAVE_ASYNC
__device__ __forceinline__ void loadA_async(const _Float16* __restrict__ Ae,
                                            int Kd, int k0, int tid,
                                            _Float16* __restrict__ As) {
#pragma unroll
  for (int i = 0; i < 2; ++i) {
    const int idx = tid + i * 256;
    const int r = idx >> 2, kq = (idx & 3) * 8;
    __builtin_amdgcn_global_load_async_to_lds_b128(
        (GAS v4i*)(Ae + (size_t)r * Kd + k0 + kq),
        (LAS v4i*)&As[r * LS + kq], 0, 0);
  }
}
#else
__device__ __forceinline__ void loadA_regs(const _Float16* __restrict__ Ae,
                                           int Kd, int k0, int tid,
                                           v8h (&av)[2]) {
#pragma unroll
  for (int i = 0; i < 2; ++i) {
    const int idx = tid + i * 256;
    const int r = idx >> 2, kq = (idx & 3) * 8;
    av[i] = *(const v8h*)(Ae + (size_t)r * Kd + k0 + kq);
  }
}
__device__ __forceinline__ void storeA_lds(_Float16* __restrict__ As, int tid,
                                           const v8h (&av)[2]) {
#pragma unroll
  for (int i = 0; i < 2; ++i) {
    const int idx = tid + i * 256;
    const int r = idx >> 2, kq = (idx & 3) * 8;
    *(v8h*)&As[r * LS + kq] = av[i];
  }
}
#endif

template <bool RELU, typename OutT>
__global__ __launch_bounds__(256) void gemm_kernel(
    const _Float16* __restrict__ A,   // [E][Cq][Kd] f16
    const float*    __restrict__ Bm,  // [E][Kd][Nd] f32
    const float*    __restrict__ bias,// [E][Nd]
    OutT*           __restrict__ Out, // [E][Cq][Nd]
    int Nd, int Kd) {
  const int e   = blockIdx.z;
  const int nb  = blockIdx.x * BN;
  const int mb  = blockIdx.y * BM;
  const int tid = threadIdx.x;
  const int lane = tid & 31;
  const int wave = tid >> 5;
  const int wm = wave >> 1;            // 0..3
  const int wn = wave & 1;             // 0..1
  const int l15 = lane & 15;
  const int lh  = lane >> 4;

  __shared__ __align__(16) _Float16 As2[2][BM * LS];   // [row][k]
  __shared__ __align__(16) _Float16 Bs2[2][BN * LS];   // [col][k] transposed

  const _Float16* Ae = A  + (size_t)e * Cq * Kd + (size_t)mb * Kd;
  const float*    Be = Bm + (size_t)e * Kd * Nd + nb;
  const float*    be = bias + (size_t)e * Nd;

  __builtin_prefetch(be + nb, 0, 1);   // global_prefetch for epilogue bias

  v8f acc[2][4];
#pragma unroll
  for (int tm = 0; tm < 2; ++tm)
#pragma unroll
    for (int tn = 0; tn < 4; ++tn)
#pragma unroll
      for (int i = 0; i < 8; ++i) acc[tm][tn][i] = 0.f;

  float4 bv[4];
#if !HAVE_ASYNC
  v8h av[2];
#endif

  // ------- prologue: fill buffer 0 for k0 = 0 -------
  loadB_regs(Be, Nd, 0, tid, bv);
#if HAVE_ASYNC
  loadA_async(Ae, Kd, 0, tid, As2[0]);
#else
  loadA_regs(Ae, Kd, 0, tid, av);
  storeA_lds(As2[0], tid, av);
#endif
  storeB_lds(Bs2[0], tid, bv);
#if HAVE_ASYNC
  __builtin_amdgcn_s_wait_asynccnt(0);
#endif
  __syncthreads();

  const int steps = Kd / BK;
  for (int s = 0; s < steps; ++s) {
    const int cur = s & 1, nxt = cur ^ 1;
    const bool more = (s + 1) < steps;
    const _Float16* Ac = As2[cur];
    const _Float16* Bc = Bs2[cur];

    // issue next tile's loads before computing (latency overlap)
    if (more) {
      loadB_regs(Be, Nd, (s + 1) * BK, tid, bv);
#if HAVE_ASYNC
      loadA_async(Ae, Kd, (s + 1) * BK, tid, As2[nxt]);
#else
      loadA_regs(Ae, Kd, (s + 1) * BK, tid, av);
#endif
    }

    // --- fragments per documented gfx1250 layouts ---
    v16h a[2], b[4];
#pragma unroll
    for (int tm = 0; tm < 2; ++tm) {
      const int r = wm * 32 + tm * 16 + l15;
      // lanes 0-15: K 0-7 then 16-23 ; lanes 16-31: K 8-15 then 24-31
      const v8h lo = *(const v8h*)&Ac[r * LS + lh * 8];
      const v8h hi = *(const v8h*)&Ac[r * LS + lh * 8 + 16];
      a[tm] = __builtin_shufflevector(lo, hi, 0, 1, 2, 3, 4, 5, 6, 7,
                                      8, 9, 10, 11, 12, 13, 14, 15);
    }
#pragma unroll
    for (int tn = 0; tn < 4; ++tn) {
      const int c = wn * 64 + tn * 16 + l15;
      // lanes 0-15: K 0-7 then 16-23 ; lanes 16-31: K 8-15 then 24-31
      const v8h lo = *(const v8h*)&Bc[c * LS + lh * 8];
      const v8h hi = *(const v8h*)&Bc[c * LS + lh * 8 + 16];
      b[tn] = __builtin_shufflevector(lo, hi, 0, 1, 2, 3, 4, 5, 6, 7,
                                      8, 9, 10, 11, 12, 13, 14, 15);
    }
#pragma unroll
    for (int tm = 0; tm < 2; ++tm)
#pragma unroll
      for (int tn = 0; tn < 4; ++tn)
        acc[tm][tn] = __builtin_amdgcn_wmma_f32_16x16x32_f16(
            false, a[tm], false, b[tn], (short)0, acc[tm][tn], false, false);

    // land next tile into the other buffer
    if (more) {
      storeB_lds(Bs2[nxt], tid, bv);
#if !HAVE_ASYNC
      storeA_lds(As2[nxt], tid, av);
#endif
#if HAVE_ASYNC
      __builtin_amdgcn_s_wait_asynccnt(0);  // ASYNCcnt not covered by barrier
#endif
    }
    __syncthreads();
  }

  // --- epilogue: bias (+relu), store ---
  OutT* Oe = Out + (size_t)e * Cq * Nd;
#pragma unroll
  for (int tm = 0; tm < 2; ++tm) {
    const int rb = mb + wm * 32 + tm * 16 + lh * 8;  // VGPR i -> row rb+i
#pragma unroll
    for (int tn = 0; tn < 4; ++tn) {
      const int col = nb + wn * 64 + tn * 16 + l15;
      const float bvv = be[col];
#pragma unroll
      for (int i = 0; i < 8; ++i) {
        float v = acc[tm][tn][i] + bvv;
        if (RELU) v = fmaxf(v, 0.f);
        Oe[(size_t)(rb + i) * Nd + col] = (OutT)v;
      }
    }
  }
}

// ---------------------------------------------------------------- combine ---
__global__ __launch_bounds__(256) void combine_kernel(
    const float* __restrict__ y,      // [E*C][M] f32
    const int* __restrict__ slots, const float* __restrict__ gates,
    float* __restrict__ out) {
  const int n = blockIdx.x;
  const int s0 = slots[n * 2 + 0], s1 = slots[n * 2 + 1];
  const float g0 = gates[n * 2 + 0], g1 = gates[n * 2 + 1];
  float4* o = (float4*)(out + (size_t)n * Mq);
  const int i = threadIdx.x;          // 256 threads == M/4
  float4 r; r.x = r.y = r.z = r.w = 0.f;
  if (s0 >= 0) {
    const float4 v = ((const float4*)(y + (size_t)s0 * Mq))[i];
    r.x += g0 * v.x; r.y += g0 * v.y; r.z += g0 * v.z; r.w += g0 * v.w;
  }
  if (s1 >= 0) {
    const float4 v = ((const float4*)(y + (size_t)s1 * Mq))[i];
    r.x += g1 * v.x; r.y += g1 * v.y; r.z += g1 * v.z; r.w += g1 * v.w;
  }
  o[i] = r;
}

// ----------------------------------------------------------------- launch ---
extern "C" void kernel_launch(void* const* d_in, const int* in_sizes, int n_in,
                              void* d_out, int out_size, void* d_ws,
                              size_t ws_size, hipStream_t stream) {
  (void)in_sizes; (void)n_in; (void)out_size; (void)ws_size;
  const float* x  = (const float*)d_in[0];   // [B,S,M]
  const float* wg = (const float*)d_in[1];   // [M,E]
  const float* w1 = (const float*)d_in[2];   // [E,M,H]
  const float* b1 = (const float*)d_in[3];   // [E,H]
  const float* w2 = (const float*)d_in[4];   // [E,H,M]
  const float* b2 = (const float*)d_in[5];   // [E,M]
  float* out = (float*)d_out;

  char* ws = (char*)d_ws;
  size_t off = 0;
  auto alloc = [&](size_t bytes) -> void* {
    void* p = ws + off;
    off = (off + bytes + 255) & ~(size_t)255;
    return p;
  };
  int*      top_i = (int*)alloc((size_t)Nq * 2 * sizeof(int));
  float*    gates = (float*)alloc((size_t)Nq * 2 * sizeof(float));
  int*      slots = (int*)alloc((size_t)Nq * 2 * sizeof(int));
  _Float16* disp  = (_Float16*)alloc((size_t)Eq * Cq * Mq * 2);  // 32 MB
  _Float16* hbuf  = (_Float16*)alloc((size_t)Eq * Cq * Hq * 2);  // 64 MB
  float*    ybuf  = (float*)alloc((size_t)Eq * Cq * Mq * 4);     // 64 MB

  gate_kernel<<<Nq, 128, 0, stream>>>(x, wg, top_i, gates);
  scan_kernel<<<1, 1024, 0, stream>>>(top_i, slots);
  dispatch_kernel<<<Nq * Kq, 256, 0, stream>>>(x, slots, disp);
  gemm_kernel<true, _Float16>
      <<<dim3(Hq / BN, Cq / BM, Eq), 256, 0, stream>>>(disp, w1, b1, hbuf, Hq, Mq);
  gemm_kernel<false, float>
      <<<dim3(Mq / BN, Cq / BM, Eq), 256, 0, stream>>>(hbuf, w2, b2, ybuf, Mq, Hq);
  combine_kernel<<<Nq, 256, 0, stream>>>(ybuf, slots, gates, out);
}
